// LSTMTagger_27152783245312
// MI455X (gfx1250) — compile-verified, matrix operand order checked by reading, not run on hardware
//
#include <hip/hip_runtime.h>
#include <hip/hip_bf16.h>
#include <stddef.h>

// ---------------------------------------------------------------------------
// LSTM tagger on MI455X (gfx1250, wave32).
//   S=4096 words, L=16 chars, EC=128, HC=256, EW=512, HW=512, T=64.
// WMMA bf16 GEMMs for all batched matmuls (register-blocked 16x64 strip per
// wave, shared B panel staged in LDS via async global->LDS copies);
// persistent single-workgroup VALU scans for the serial recurrences with
// h-state in LDS and recurrent weights streamed from L2 (bf16).
// ---------------------------------------------------------------------------

#define S_WORDS   4096
#define L_CHARS   16
#define EC_DIM    128
#define HC_DIM    256
#define EW_DIM    512
#define HW_DIM    512
#define T_TAGS    64
#define NCHARS    (S_WORDS * L_CHARS)   // 65536
#define AUG_DIM   (EW_DIM + HC_DIM)     // 768

typedef __attribute__((ext_vector_type(16))) __bf16 v16bf;
typedef __attribute__((ext_vector_type(8)))  float  v8f;
typedef __attribute__((ext_vector_type(4)))  int    v4i;

struct __attribute__((aligned(16))) U4 { unsigned x, y, z, w; };

union Frag { v16bf v; U4 q[2]; };

#if __has_builtin(__builtin_amdgcn_global_load_async_to_lds_b128)
#define HAVE_ASYNC_LDS 1
#else
#define HAVE_ASYNC_LDS 0
#endif

#if __has_builtin(__builtin_amdgcn_s_wait_asynccnt)
#define WAIT_ASYNCCNT0() __builtin_amdgcn_s_wait_asynccnt(0)
#else
#define WAIT_ASYNCCNT0() asm volatile("s_wait_asynccnt 0x0" ::: "memory")
#endif

// float -> bf16 (round to nearest even), bf16 -> float
static __device__ __forceinline__ unsigned short f2bf(float f) {
    unsigned u = __float_as_uint(f);
    u += 0x7FFFu + ((u >> 16) & 1u);
    return (unsigned short)(u >> 16);
}
static __device__ __forceinline__ float bf2f(unsigned h) {
    return __uint_as_float((h & 0xFFFFu) << 16);
}
static __device__ __forceinline__ float sigmoidf_fast(float x) {
    return 1.0f / (1.0f + __expf(-x));
}

// ---------------------------------------------------------------------------
// Generic f32 -> bf16 conversion (weights)
// ---------------------------------------------------------------------------
__global__ void k_cvt_bf16(unsigned short* __restrict__ dst,
                           const float* __restrict__ src, int n) {
    int i = blockIdx.x * blockDim.x + threadIdx.x;
    if (i < n) dst[i] = f2bf(src[i]);
}

// ---------------------------------------------------------------------------
// Char-embedding gather -> bf16 activation matrix  Xc[65536, 128]
// ---------------------------------------------------------------------------
__global__ void k_gather_chars(unsigned short* __restrict__ Xc,
                               const int* __restrict__ word_chars,
                               const float* __restrict__ char_emb) {
    int idx = blockIdx.x * blockDim.x + threadIdx.x;       // over 65536*128
    if (idx >= NCHARS * EC_DIM) return;
    int i = idx >> 7;          // char position
    int e = idx & 127;         // embed dim
    Xc[idx] = f2bf(char_emb[(size_t)word_chars[i] * EC_DIM + e]);
}

// ---------------------------------------------------------------------------
// Aug matrix: [word_emb[sentence[s]] | word_rep[s]] -> bf16 [4096, 768]
// ---------------------------------------------------------------------------
__global__ void k_build_aug(unsigned short* __restrict__ aug,
                            const int* __restrict__ sentence,
                            const float* __restrict__ word_emb,
                            const float* __restrict__ word_rep) {
    int idx = blockIdx.x * blockDim.x + threadIdx.x;       // over 4096*768
    if (idx >= S_WORDS * AUG_DIM) return;
    int s = idx / AUG_DIM;
    int e = idx - s * AUG_DIM;
    float v = (e < EW_DIM)
                  ? word_emb[(size_t)sentence[s] * EW_DIM + e]
                  : word_rep[(size_t)s * HC_DIM + (e - EW_DIM)];
    aug[idx] = f2bf(v);
}

// ---------------------------------------------------------------------------
// WMMA bf16 GEMM:  C[M,N] (f32) = A[M,K] * B[N,K]^T + bias1[n] + bias2[n]
//
// Block = 256 threads = 8 waves, output tile 128 (M) x 64 (N).
// The 64-row B panel (shared by all 8 waves) is staged to LDS once via
// GLOBAL_LOAD_ASYNC_TO_LDS_B128 (ASYNCcnt) with a 16B/row skew so the
// 16-lane ds_load_b128 fragment reads are bank-conflict-free.
// Each wave holds one A fragment in VGPRs and reuses it across 4 WMMA
// accumulator tiles (16x64 strip), K consumed in chunks of 32.
// Fragment layout per CDNA5 ISA 7.12.2 (16-bit A 16x32; f32 C/D 16x16):
//   lane = half*16 + r16; A/B: row r16, k-pairs at {half*8..+7, 16+half*8..+7}
//   C/D:  column r16, VGPR v holds row m = v + 8*half.
// Requires M % 128 == 0, N % 64 == 0, K % 32 == 0 (true for all call sites).
// ---------------------------------------------------------------------------
__global__ void __launch_bounds__(256)
k_gemm_bf16(float* __restrict__ C,
            const unsigned short* __restrict__ A,
            const unsigned short* __restrict__ B,
            const float* __restrict__ bias1,
            const float* __restrict__ bias2,
            int M, int N, int K) {
    extern __shared__ unsigned short panel[];   // [64][K + 8] bf16, skewed rows

    const int t    = threadIdx.x;
    const int lane = t & 31;
    const int wave = t >> 5;
    const int nstrips = N >> 6;
    const int bm = blockIdx.x / nstrips;        // 128-row block index
    const int bn = blockIdx.x - bm * nstrips;   // 64-col strip index
    const int ldp = K + 8;                      // skewed LDS row stride (elems)

    // ---- Stage B strip [64 x K] -> LDS (async copy path) ----
    const int chunksPerRow = K >> 3;            // 16B chunks per row
    const int totalChunks  = chunksPerRow << 6; // x64 rows (multiple of 256)
    for (int ch = t; ch < totalChunks; ch += 256) {
        int row = ch / chunksPerRow;
        int col = (ch - row * chunksPerRow) << 3;            // element offset
        const unsigned short* gsrc = B + (size_t)(bn * 64 + row) * K + col;
        unsigned short*       ldst = panel + row * ldp + col;
#if HAVE_ASYNC_LDS
        __builtin_amdgcn_global_load_async_to_lds_b128(
            (__attribute__((address_space(1))) v4i*)gsrc,
            (__attribute__((address_space(3))) v4i*)ldst,
            0, 0);
#else
        *(U4*)ldst = *(const U4*)gsrc;
#endif
    }
#if HAVE_ASYNC_LDS
    WAIT_ASYNCCNT0();
#endif
    __syncthreads();

    // ---- K loop: 1 A fragment reused across 4 B fragments / 4 acc tiles ----
    const int r16  = lane & 15;
    const int half = lane >> 4;
    const char* arow = (const char*)(A + (size_t)(bm * 128 + wave * 16 + r16) * K);
    const unsigned short* prow = panel + r16 * ldp;   // B row r16 of subtile 0

    v8f acc0 = {}, acc1 = {}, acc2 = {}, acc3 = {};
    for (int kk = 0; kk < K; kk += 32) {
        Frag fa;
        const char* pa = arow + (size_t)kk * 2;
        fa.q[0] = *(const U4*)(pa + half * 16);
        fa.q[1] = *(const U4*)(pa + 32 + half * 16);

        Frag fb0, fb1, fb2, fb3;
        const unsigned short* p0 = prow + kk + half * 8;
        fb0.q[0] = *(const U4*)(p0);
        fb0.q[1] = *(const U4*)(p0 + 16);
        const unsigned short* p1 = p0 + 16 * ldp;
        fb1.q[0] = *(const U4*)(p1);
        fb1.q[1] = *(const U4*)(p1 + 16);
        const unsigned short* p2 = p1 + 16 * ldp;
        fb2.q[0] = *(const U4*)(p2);
        fb2.q[1] = *(const U4*)(p2 + 16);
        const unsigned short* p3 = p2 + 16 * ldp;
        fb3.q[0] = *(const U4*)(p3);
        fb3.q[1] = *(const U4*)(p3 + 16);

        acc0 = __builtin_amdgcn_wmma_f32_16x16x32_bf16(
            false, fa.v, false, fb0.v, (short)0, acc0, false, false);
        acc1 = __builtin_amdgcn_wmma_f32_16x16x32_bf16(
            false, fa.v, false, fb1.v, (short)0, acc1, false, false);
        acc2 = __builtin_amdgcn_wmma_f32_16x16x32_bf16(
            false, fa.v, false, fb2.v, (short)0, acc2, false, false);
        acc3 = __builtin_amdgcn_wmma_f32_16x16x32_bf16(
            false, fa.v, false, fb3.v, (short)0, acc3, false, false);
    }

    // ---- Epilogue: bias + store ----
    const int nc0 = bn * 64 + r16;
    float bs0 = 0.f, bs1 = 0.f, bs2 = 0.f, bs3 = 0.f;
    if (bias1) {
        bs0 += bias1[nc0]; bs1 += bias1[nc0 + 16];
        bs2 += bias1[nc0 + 32]; bs3 += bias1[nc0 + 48];
    }
    if (bias2) {
        bs0 += bias2[nc0]; bs1 += bias2[nc0 + 16];
        bs2 += bias2[nc0 + 32]; bs3 += bias2[nc0 + 48];
    }
#pragma unroll
    for (int r = 0; r < 8; ++r) {
        int m = bm * 128 + wave * 16 + r + 8 * half;
        float* crow = C + (size_t)m * N + nc0;
        crow[0]  = acc0[r] + bs0;
        crow[16] = acc1[r] + bs1;
        crow[32] = acc2[r] + bs2;
        crow[48] = acc3[r] + bs3;
    }
}

// ---------------------------------------------------------------------------
// Char LSTM scan: 65536 serial steps, H=256, gates=1024.
// One workgroup, 1024 threads; thread t owns gate row t.
// Whh (bf16, 512 KB) streamed per step -> stays hot in L2 (192 MB).
// h-state in LDS (same-address reads broadcast across the wave).
// ---------------------------------------------------------------------------
__global__ void __launch_bounds__(1024)
k_char_scan(const float* __restrict__ xg,            // [65536, 1024] f32
            const unsigned short* __restrict__ Whh,  // [1024, 256] bf16
            float* __restrict__ word_rep) {          // [4096, 256] f32
    __shared__ float2 hls[HC_DIM / 2];   // h as 256 f32
    __shared__ float  gates[1024];

    const int t = threadIdx.x;
    float c = 0.0f;
    if (t < HC_DIM / 2) { float2 z; z.x = 0.0f; z.y = 0.0f; hls[t] = z; }
    __syncthreads();

    const U4* wrow = (const U4*)(Whh + (size_t)t * HC_DIM);  // 32 x u4 = 256 bf16

    for (int s = 0; s < NCHARS; ++s) {
        float acc = xg[(size_t)s * 1024 + t];
        __builtin_prefetch(xg + (size_t)(s + 1) * 1024 + t, 0, 0);
#pragma unroll 8
        for (int j = 0; j < 32; ++j) {
            U4 w = wrow[j];
            float2 h0 = hls[4 * j + 0], h1 = hls[4 * j + 1];
            float2 h2 = hls[4 * j + 2], h3 = hls[4 * j + 3];
            acc += bf2f(w.x) * h0.x + bf2f(w.x >> 16) * h0.y;
            acc += bf2f(w.y) * h1.x + bf2f(w.y >> 16) * h1.y;
            acc += bf2f(w.z) * h2.x + bf2f(w.z >> 16) * h2.y;
            acc += bf2f(w.w) * h3.x + bf2f(w.w >> 16) * h3.y;
        }
        // gate order i(0..255) f(256..511) g(512..767) o(768..1023)
        gates[t] = (t >= 512 && t < 768) ? tanhf(acc) : sigmoidf_fast(acc);
        __syncthreads();
        if (t < HC_DIM) {
            float i = gates[t], f = gates[256 + t];
            float g = gates[512 + t], o = gates[768 + t];
            c = f * c + i * g;
            float h = o * tanhf(c);
            ((float*)hls)[t] = h;
            if ((s & (L_CHARS - 1)) == (L_CHARS - 1))
                word_rep[(size_t)(s >> 4) * HC_DIM + t] = h;
        }
        __syncthreads();
    }
}

// ---------------------------------------------------------------------------
// Sentence LSTM scan: 4096 serial steps, H=512, gates=2048.
// 1024 threads, thread t owns rows t (i/f) and t+1024 (g/o).
// Emits hidden states directly as bf16 for the logits WMMA GEMM.
// ---------------------------------------------------------------------------
__global__ void __launch_bounds__(1024)
k_sent_scan(const float* __restrict__ xg,            // [4096, 2048] f32
            const unsigned short* __restrict__ Whh,  // [2048, 512] bf16
            unsigned short* __restrict__ hs_out) {   // [4096, 512] bf16
    __shared__ float2 hls[HW_DIM / 2];   // h as 512 f32
    __shared__ float  gates[2048];

    const int t = threadIdx.x;
    const int r0 = t, r1 = t + 1024;
    float c = 0.0f;
    if (t < HW_DIM / 2) { float2 z; z.x = 0.0f; z.y = 0.0f; hls[t] = z; }
    __syncthreads();

    const U4* w0 = (const U4*)(Whh + (size_t)r0 * HW_DIM);  // 64 x u4 each
    const U4* w1 = (const U4*)(Whh + (size_t)r1 * HW_DIM);

    for (int s = 0; s < S_WORDS; ++s) {
        float a0 = xg[(size_t)s * 2048 + r0];
        float a1 = xg[(size_t)s * 2048 + r1];
        __builtin_prefetch(xg + (size_t)(s + 1) * 2048 + r0, 0, 0);
#pragma unroll 4
        for (int j = 0; j < 64; ++j) {
            U4 wa = w0[j];
            U4 wb = w1[j];
            float2 h0 = hls[4 * j + 0], h1 = hls[4 * j + 1];
            float2 h2 = hls[4 * j + 2], h3 = hls[4 * j + 3];
            a0 += bf2f(wa.x) * h0.x + bf2f(wa.x >> 16) * h0.y;
            a0 += bf2f(wa.y) * h1.x + bf2f(wa.y >> 16) * h1.y;
            a0 += bf2f(wa.z) * h2.x + bf2f(wa.z >> 16) * h2.y;
            a0 += bf2f(wa.w) * h3.x + bf2f(wa.w >> 16) * h3.y;
            a1 += bf2f(wb.x) * h0.x + bf2f(wb.x >> 16) * h0.y;
            a1 += bf2f(wb.y) * h1.x + bf2f(wb.y >> 16) * h1.y;
            a1 += bf2f(wb.z) * h2.x + bf2f(wb.z >> 16) * h2.y;
            a1 += bf2f(wb.w) * h3.x + bf2f(wb.w >> 16) * h3.y;
        }
        // i(0..511) f(512..1023) g(1024..1535) o(1536..2047)
        gates[r0] = sigmoidf_fast(a0);                            // i or f
        gates[r1] = (r1 < 1536) ? tanhf(a1) : sigmoidf_fast(a1);  // g or o
        __syncthreads();
        if (t < HW_DIM) {
            float i = gates[t], f = gates[512 + t];
            float g = gates[1024 + t], o = gates[1536 + t];
            c = f * c + i * g;
            float h = o * tanhf(c);
            ((float*)hls)[t] = h;
            hs_out[(size_t)s * HW_DIM + t] = f2bf(h);
        }
        __syncthreads();
    }
}

// ---------------------------------------------------------------------------
// log_softmax over rows of 64: one wave32 per row, 2 elements per lane.
// ---------------------------------------------------------------------------
__global__ void k_logsoftmax(float* __restrict__ out,
                             const float* __restrict__ logits, int rows) {
    const int lane = threadIdx.x & 31;
    const int wave = threadIdx.x >> 5;
    const int row  = blockIdx.x * (blockDim.x >> 5) + wave;
    if (row >= rows) return;
    const float* L = logits + (size_t)row * T_TAGS;
    float a = L[lane], b = L[lane + 32];
    float m = fmaxf(a, b);
    for (int d = 16; d > 0; d >>= 1) m = fmaxf(m, __shfl_xor(m, d, 32));
    float s = __expf(a - m) + __expf(b - m);
    for (int d = 16; d > 0; d >>= 1) s += __shfl_xor(s, d, 32);
    float lse = m + __logf(s);
    out[(size_t)row * T_TAGS + lane]      = a - lse;
    out[(size_t)row * T_TAGS + lane + 32] = b - lse;
}

// ---------------------------------------------------------------------------
// Orchestration
// ---------------------------------------------------------------------------
extern "C" void kernel_launch(void* const* d_in, const int* in_sizes, int n_in,
                              void* d_out, int out_size, void* d_ws, size_t ws_size,
                              hipStream_t stream) {
    const int*   word_chars = (const int*)d_in[0];
    const int*   sentence   = (const int*)d_in[1];
    const float* char_emb   = (const float*)d_in[2];
    const float* word_emb   = (const float*)d_in[3];
    const float* Wih_c      = (const float*)d_in[4];
    const float* Whh_c      = (const float*)d_in[5];
    const float* bih_c      = (const float*)d_in[6];
    const float* bhh_c      = (const float*)d_in[7];
    const float* Wih_s      = (const float*)d_in[8];
    const float* Whh_s      = (const float*)d_in[9];
    const float* bih_s      = (const float*)d_in[10];
    const float* bhh_s      = (const float*)d_in[11];
    const float* W_tag      = (const float*)d_in[12];
    const float* b_tag      = (const float*)d_in[13];
    float* out = (float*)d_out;

    char* ws = (char*)d_ws;
    size_t off = 0;
    auto carve = [&](size_t bytes) -> void* {
        void* p = ws + off;
        off = (off + bytes + 255) & ~(size_t)255;
        return p;
    };
    unsigned short* wsXc    = (unsigned short*)carve((size_t)NCHARS * EC_DIM * 2);
    unsigned short* wsWihC  = (unsigned short*)carve((size_t)4 * HC_DIM * EC_DIM * 2);
    unsigned short* wsWhhC  = (unsigned short*)carve((size_t)4 * HC_DIM * HC_DIM * 2);
    unsigned short* wsWihS  = (unsigned short*)carve((size_t)4 * HW_DIM * AUG_DIM * 2);
    unsigned short* wsWhhS  = (unsigned short*)carve((size_t)4 * HW_DIM * HW_DIM * 2);
    unsigned short* wsWtag  = (unsigned short*)carve((size_t)T_TAGS * HW_DIM * 2);
    float*          wsXgC   = (float*)carve((size_t)NCHARS * 4 * HC_DIM * 4);
    float*          wsWrep  = (float*)carve((size_t)S_WORDS * HC_DIM * 4);
    unsigned short* wsAug   = (unsigned short*)carve((size_t)S_WORDS * AUG_DIM * 2);
    float*          wsXgS   = (float*)carve((size_t)S_WORDS * 4 * HW_DIM * 4);
    unsigned short* wsHsS   = (unsigned short*)carve((size_t)S_WORDS * HW_DIM * 2);
    float*          wsLogit = (float*)carve((size_t)S_WORDS * T_TAGS * 4);

    const int TB = 256;
    auto cdiv = [](int a, int b) { return (a + b - 1) / b; };
    auto panel_bytes = [](int K) { return (size_t)64 * (K + 8) * 2; };

    // 1) Weight conversions to bf16
    k_cvt_bf16<<<cdiv(4 * HC_DIM * EC_DIM, TB), TB, 0, stream>>>(wsWihC, Wih_c, 4 * HC_DIM * EC_DIM);
    k_cvt_bf16<<<cdiv(4 * HC_DIM * HC_DIM, TB), TB, 0, stream>>>(wsWhhC, Whh_c, 4 * HC_DIM * HC_DIM);
    k_cvt_bf16<<<cdiv(4 * HW_DIM * AUG_DIM, TB), TB, 0, stream>>>(wsWihS, Wih_s, 4 * HW_DIM * AUG_DIM);
    k_cvt_bf16<<<cdiv(4 * HW_DIM * HW_DIM, TB), TB, 0, stream>>>(wsWhhS, Whh_s, 4 * HW_DIM * HW_DIM);
    k_cvt_bf16<<<cdiv(T_TAGS * HW_DIM, TB), TB, 0, stream>>>(wsWtag, W_tag, T_TAGS * HW_DIM);

    // 2) Char embedding gather (bf16 A matrix)
    k_gather_chars<<<cdiv(NCHARS * EC_DIM, TB), TB, 0, stream>>>(wsXc, word_chars, char_emb);

    // 3) xg_c = Xc @ Wih_c^T + (bih_c + bhh_c)   [65536 x 1024], WMMA
    {
        int blocks = (NCHARS / 128) * (4 * HC_DIM / 64);   // 8192
        k_gemm_bf16<<<blocks, TB, panel_bytes(EC_DIM), stream>>>(
            wsXgC, wsXc, wsWihC, bih_c, bhh_c, NCHARS, 4 * HC_DIM, EC_DIM);
    }

    // 4) Serial char-LSTM scan (single persistent workgroup)
    k_char_scan<<<1, 1024, 0, stream>>>(wsXgC, wsWhhC, wsWrep);

    // 5) Aug = [word_emb gather | word_rep] -> bf16
    k_build_aug<<<cdiv(S_WORDS * AUG_DIM, TB), TB, 0, stream>>>(wsAug, sentence, word_emb, wsWrep);

    // 6) xg_s = Aug @ Wih_s^T + (bih_s + bhh_s)   [4096 x 2048], WMMA
    {
        int blocks = (S_WORDS / 128) * (4 * HW_DIM / 64);  // 1024
        k_gemm_bf16<<<blocks, TB, panel_bytes(AUG_DIM), stream>>>(
            wsXgS, wsAug, wsWihS, bih_s, bhh_s, S_WORDS, 4 * HW_DIM, AUG_DIM);
    }

    // 7) Serial sentence-LSTM scan (emits bf16 hidden states)
    k_sent_scan<<<1, 1024, 0, stream>>>(wsXgS, wsWhhS, wsHsS);

    // 8) logits = hs_s @ W_tag^T + b_tag   [4096 x 64], WMMA
    {
        int blocks = (S_WORDS / 128) * (T_TAGS / 64);      // 32
        k_gemm_bf16<<<blocks, TB, panel_bytes(HW_DIM), stream>>>(
            wsLogit, wsHsS, wsWtag, b_tag, (const float*)nullptr,
            S_WORDS, T_TAGS, HW_DIM);
    }

    // 9) log_softmax -> d_out
    k_logsoftmax<<<cdiv(S_WORDS, TB / 32), TB, 0, stream>>>(out, wsLogit, S_WORDS);

    (void)in_sizes; (void)n_in; (void)out_size; (void)ws_size;
}
